// MultiHeadAttention_1211180778462
// MI455X (gfx1250) — matrix operand order repaired and checked by reading, not compiled, not run on hardware
//
#include <hip/hip_runtime.h>

#define B_ 2
#define S_ 2048
#define D_ 1024
#define H_ 16
#define DH_ 64

typedef __bf16 bf16;
typedef __bf16 v16bf __attribute__((ext_vector_type(16)));
typedef __bf16 v8bf  __attribute__((ext_vector_type(8)));
typedef float  v8f   __attribute__((ext_vector_type(8)));

// ---- WMMA: D = A(16x32 bf16) * B(32x16 bf16) + C(16x16 f32) ----
static __device__ inline v8f wmma_bf16(v16bf a, v16bf b, v8f c) {
    return __builtin_amdgcn_wmma_f32_16x16x32_bf16(
        false, a, false, b, (short)0, c, false, false);
}

// ---- A fragment (16x32): elems 0..7 = K[k0+8h..], 8..15 = K[k0+16+8h..] ----
static __device__ inline v16bf afrag_ld(const bf16* row, int k0, int half) {
    union { v16bf v; v8bf h[2]; } u;
    u.h[0] = *(const v8bf*)(row + k0 + half * 8);
    u.h[1] = *(const v8bf*)(row + k0 + 16 + half * 8);
    return u.v;
}
// ---- B fragment (32x16): elems 0..7 = K[k0+8h..], 8..15 = K[k0+16+8h..] ----
static __device__ inline v16bf bfrag_ld(const bf16* row, int k0, int half) {
    union { v16bf v; v8bf h[2]; } u;
    u.h[0] = *(const v8bf*)(row + k0 + half * 8);
    u.h[1] = *(const v8bf*)(row + k0 + 16 + half * 8);
    return u.v;
}

// ============================================================================
// Kernel 0: f32 -> bf16 bulk convert (8 elems / thread, vectorized)
// ============================================================================
__global__ void __launch_bounds__(256) cvt_kernel(
    const float* __restrict__ src, bf16* __restrict__ dst, int n8) {
    const int i = blockIdx.x * 256 + threadIdx.x;
    if (i < n8) {
        const float4* s4 = (const float4*)src;
        const float4 a = s4[2 * i];
        const float4 b = s4[2 * i + 1];
        v8bf o;
        o[0] = (bf16)a.x; o[1] = (bf16)a.y; o[2] = (bf16)a.z; o[3] = (bf16)a.w;
        o[4] = (bf16)b.x; o[5] = (bf16)b.y; o[6] = (bf16)b.z; o[7] = (bf16)b.w;
        *(v8bf*)(dst + (size_t)i * 8) = o;
    }
}

// ============================================================================
// Kernel 1: QKV projection. Xbf[B*S, D] x Wqkv_bf[3D, D]^T -> Q/K [B,H,S,DH],
// Vt [B,H,DH,S]. Each wave: 32(M) x 64(N), K-steps of 32, double-buffered.
// ============================================================================
__global__ void __launch_bounds__(128) qkv_kernel(
    const bf16* __restrict__ Xbf, const bf16* __restrict__ Wq,
    bf16* __restrict__ Q, bf16* __restrict__ K, bf16* __restrict__ Vt) {
    const int lane = threadIdx.x & 31;
    const int wave = threadIdx.x >> 5;
    const int half = lane >> 4;
    const int l15  = lane & 15;

    const int NSTRIPS = (3 * D_) / 64;              // 48
    const int wid = blockIdx.x * 4 + wave;          // 0..6143
    const int m0  = (wid / NSTRIPS) * 32;
    const int n0  = (wid % NSTRIPS) * 64;

    const bf16* x0 = Xbf + (size_t)(m0 + l15) * D_;
    const bf16* x1 = x0 + (size_t)16 * D_;
    const bf16* w0 = Wq + (size_t)(n0 + l15) * D_;

    v8f acc[8] = {};
    v16bf a0 = afrag_ld(x0, 0, half);
    v16bf a1 = afrag_ld(x1, 0, half);
    v16bf b0 = bfrag_ld(w0, 0, half);
    v16bf b1 = bfrag_ld(w0 + (size_t)16 * D_, 0, half);
    v16bf b2 = bfrag_ld(w0 + (size_t)32 * D_, 0, half);
    v16bf b3 = bfrag_ld(w0 + (size_t)48 * D_, 0, half);

    for (int k0 = 0; k0 < D_; k0 += 32) {
        const int kn = (k0 + 32) & (D_ - 1);        // wraps on last iter (harmless)
        v16bf na0 = afrag_ld(x0, kn, half);
        v16bf na1 = afrag_ld(x1, kn, half);
        v16bf nb0 = bfrag_ld(w0, kn, half);
        v16bf nb1 = bfrag_ld(w0 + (size_t)16 * D_, kn, half);
        v16bf nb2 = bfrag_ld(w0 + (size_t)32 * D_, kn, half);
        v16bf nb3 = bfrag_ld(w0 + (size_t)48 * D_, kn, half);

        acc[0] = wmma_bf16(a0, b0, acc[0]);
        acc[1] = wmma_bf16(a0, b1, acc[1]);
        acc[2] = wmma_bf16(a0, b2, acc[2]);
        acc[3] = wmma_bf16(a0, b3, acc[3]);
        acc[4] = wmma_bf16(a1, b0, acc[4]);
        acc[5] = wmma_bf16(a1, b1, acc[5]);
        acc[6] = wmma_bf16(a1, b2, acc[6]);
        acc[7] = wmma_bf16(a1, b3, acc[7]);

        a0 = na0; a1 = na1; b0 = nb0; b1 = nb1; b2 = nb2; b3 = nb3;
    }

    // scatter into Q / K / Vt: output channel e = h*192 + c
#pragma unroll
    for (int mt = 0; mt < 2; ++mt) {
#pragma unroll
        for (int nt = 0; nt < 4; ++nt) {
            const int e = n0 + nt * 16 + l15;
            const int h = e / (3 * DH_);
            const int c = e % (3 * DH_);
            const v8f* pacc = &acc[mt * 4 + nt];
#pragma unroll
            for (int r = 0; r < 8; ++r) {
                const int m = m0 + mt * 16 + r + 8 * half;
                const int b = m / S_;
                const int s = m % S_;
                const size_t bh = (size_t)(b * H_ + h);
                const bf16 val = (bf16)((*pacc)[r]);
                if (c < DH_)          Q [(bh * S_ + s) * DH_ + c]             = val;
                else if (c < 2*DH_)   K [(bh * S_ + s) * DH_ + (c - DH_)]     = val;
                else                  Vt[(bh * DH_ + (c - 2*DH_)) * S_ + s]   = val;
            }
        }
    }
}

// ============================================================================
// Kernel 2: streaming-softmax attention. One wave per 32-query tile (2x16)
// of one (b,h); key blocks of 32. K/V fragments are loaded once per block
// and reused by both query sub-tiles (halves L2 K/V traffic). Row max/sum/
// rescale are pure-register 16-lane shuffle reductions; only the P
// transposes (D-layout -> A-layout) go through LDS.
// ============================================================================
__global__ void __launch_bounds__(128) attn_kernel(
    const bf16* __restrict__ Q, const bf16* __restrict__ K,
    const bf16* __restrict__ Vt, bf16* __restrict__ vals) {
    __shared__ __align__(64) bf16 s_p[4][2][16][32];   // [wave][qt][row][col]

    const int lane = threadIdx.x & 31;
    const int wave = threadIdx.x >> 5;
    const int half = lane >> 4;
    const int l15  = lane & 15;

    const int wid = blockIdx.x * 4 + wave;          // 32-query tile id, 0..2047
    const int QT  = S_ / 32;                        // 64
    const int bh  = wid / QT;                       // 0..31
    const int q0  = (wid % QT) * 32;

    const bf16* Qb = Q  + (size_t)bh * S_ * DH_;
    const bf16* Kb = K  + (size_t)bh * S_ * DH_;
    const bf16* Vb = Vt + (size_t)bh * DH_ * S_;

    // preload Q A-frags for both 16-row sub-tiles, pre-scaled by 1/sqrt(DH)
    v16bf qa[2][2];
#pragma unroll
    for (int qt = 0; qt < 2; ++qt) {
        const bf16* qrow = Qb + (size_t)(q0 + qt * 16 + l15) * DH_;
        v16bf r0 = afrag_ld(qrow, 0, half);
        v16bf r1 = afrag_ld(qrow, 32, half);
#pragma unroll
        for (int i = 0; i < 16; ++i) {
            qa[qt][0][i] = (bf16)((float)r0[i] * 0.125f);
            qa[qt][1][i] = (bf16)((float)r1[i] * 0.125f);
        }
    }

    v8f o[2][4] = {};
    v8f m_run[2], l_run[2];
#pragma unroll
    for (int qt = 0; qt < 2; ++qt)
#pragma unroll
        for (int r = 0; r < 8; ++r) { m_run[qt][r] = -1e30f; l_run[qt][r] = 0.0f; }

    for (int kb = 0; kb < S_; kb += 32) {
        // ---- K / V fragments: loaded once, used by both query sub-tiles ----
        const bf16* kr0 = Kb + (size_t)(kb + l15) * DH_;
        const bf16* kr1 = kr0 + (size_t)16 * DH_;
        v16bf kf00 = bfrag_ld(kr0, 0, half);
        v16bf kf01 = bfrag_ld(kr0, 32, half);
        v16bf kf10 = bfrag_ld(kr1, 0, half);
        v16bf kf11 = bfrag_ld(kr1, 32, half);
        v16bf vf[4];
#pragma unroll
        for (int nd = 0; nd < 4; ++nd)
            vf[nd] = bfrag_ld(Vb + (size_t)(nd * 16 + l15) * S_ + kb, 0, half);

#pragma unroll
        for (int qt = 0; qt < 2; ++qt) {
            // ---- scores for this 16-query sub-tile ----
            v8f s0 = {}, s1 = {};
            s0 = wmma_bf16(qa[qt][0], kf00, s0);
            s0 = wmma_bf16(qa[qt][1], kf01, s0);
            s1 = wmma_bf16(qa[qt][0], kf10, s1);
            s1 = wmma_bf16(qa[qt][1], kf11, s1);

            // ---- in-register online softmax (all 32 lanes) ----
            v8f m_new, corr;
#pragma unroll
            for (int r = 0; r < 8; ++r) {
                float x = fmaxf(s0[r], s1[r]);
#pragma unroll
                for (int msk = 8; msk >= 1; msk >>= 1)
                    x = fmaxf(x, __shfl_xor(x, msk, 32));
                m_new[r] = fmaxf(m_run[qt][r], x);
                corr[r]  = __expf(m_run[qt][r] - m_new[r]);
            }
#pragma unroll
            for (int r = 0; r < 8; ++r) {
                const float p0 = __expf(s0[r] - m_new[r]);
                const float p1 = __expf(s1[r] - m_new[r]);
                s_p[wave][qt][r + 8 * half][l15]      = (bf16)p0;
                s_p[wave][qt][r + 8 * half][16 + l15] = (bf16)p1;
                float s = p0 + p1;
#pragma unroll
                for (int msk = 8; msk >= 1; msk >>= 1)
                    s += __shfl_xor(s, msk, 32);
                l_run[qt][r] = l_run[qt][r] * corr[r] + s;
                m_run[qt][r] = m_new[r];
            }
#pragma unroll
            for (int nd = 0; nd < 4; ++nd)
#pragma unroll
                for (int r = 0; r < 8; ++r) o[qt][nd][r] *= corr[r];

            // ---- P A-frag from LDS, then O += P @ V ----
            const v16bf pa = afrag_ld(&s_p[wave][qt][l15][0], 0, half);
#pragma unroll
            for (int nd = 0; nd < 4; ++nd)
                o[qt][nd] = wmma_bf16(pa, vf[nd], o[qt][nd]);
        }
    }

    // ---- final 1/l normalization (in-register) + store vals ----
    const int b = bh / H_;
    const int h = bh % H_;
#pragma unroll
    for (int qt = 0; qt < 2; ++qt) {
        v8f inv;
#pragma unroll
        for (int r = 0; r < 8; ++r) inv[r] = 1.0f / l_run[qt][r];
#pragma unroll
        for (int nd = 0; nd < 4; ++nd)
#pragma unroll
            for (int r = 0; r < 8; ++r) {
                const int srow = q0 + qt * 16 + r + 8 * half;
                vals[(size_t)(b * S_ + srow) * D_ + h * DH_ + nd * 16 + l15] =
                    (bf16)(o[qt][nd][r] * inv[r]);
            }
    }
}

// ============================================================================
// Kernel 3: output projection. vals[B*S, D] x Wout_bf[D, D]^T -> out (f32).
// Each wave: 32(M) x 64(N), double-buffered K-steps of 32.
// ============================================================================
__global__ void __launch_bounds__(128) out_kernel(
    const bf16* __restrict__ vals, const bf16* __restrict__ Wo,
    float* __restrict__ out) {
    const int lane = threadIdx.x & 31;
    const int wave = threadIdx.x >> 5;
    const int half = lane >> 4;
    const int l15  = lane & 15;

    const int NSTRIPS = D_ / 64;                    // 16
    const int wid = blockIdx.x * 4 + wave;          // 0..2047
    const int m0  = (wid / NSTRIPS) * 32;
    const int n0  = (wid % NSTRIPS) * 64;

    const bf16* x0 = vals + (size_t)(m0 + l15) * D_;
    const bf16* x1 = x0 + (size_t)16 * D_;
    const bf16* w0 = Wo + (size_t)(n0 + l15) * D_;

    v8f acc[8] = {};
    v16bf a0 = afrag_ld(x0, 0, half);
    v16bf a1 = afrag_ld(x1, 0, half);
    v16bf b0 = bfrag_ld(w0, 0, half);
    v16bf b1 = bfrag_ld(w0 + (size_t)16 * D_, 0, half);
    v16bf b2 = bfrag_ld(w0 + (size_t)32 * D_, 0, half);
    v16bf b3 = bfrag_ld(w0 + (size_t)48 * D_, 0, half);

    for (int k0 = 0; k0 < D_; k0 += 32) {
        const int kn = (k0 + 32) & (D_ - 1);
        v16bf na0 = afrag_ld(x0, kn, half);
        v16bf na1 = afrag_ld(x1, kn, half);
        v16bf nb0 = bfrag_ld(w0, kn, half);
        v16bf nb1 = bfrag_ld(w0 + (size_t)16 * D_, kn, half);
        v16bf nb2 = bfrag_ld(w0 + (size_t)32 * D_, kn, half);
        v16bf nb3 = bfrag_ld(w0 + (size_t)48 * D_, kn, half);

        acc[0] = wmma_bf16(a0, b0, acc[0]);
        acc[1] = wmma_bf16(a0, b1, acc[1]);
        acc[2] = wmma_bf16(a0, b2, acc[2]);
        acc[3] = wmma_bf16(a0, b3, acc[3]);
        acc[4] = wmma_bf16(a1, b0, acc[4]);
        acc[5] = wmma_bf16(a1, b1, acc[5]);
        acc[6] = wmma_bf16(a1, b2, acc[6]);
        acc[7] = wmma_bf16(a1, b3, acc[7]);

        a0 = na0; a1 = na1; b0 = nb0; b1 = nb1; b2 = nb2; b3 = nb3;
    }

#pragma unroll
    for (int mt = 0; mt < 2; ++mt)
#pragma unroll
        for (int nt = 0; nt < 4; ++nt)
#pragma unroll
            for (int r = 0; r < 8; ++r)
                out[(size_t)(m0 + mt * 16 + r + 8 * half) * D_ + n0 + nt * 16 + l15] =
                    acc[mt * 4 + nt][r];
}

// ============================================================================
extern "C" void kernel_launch(void* const* d_in, const int* in_sizes, int n_in,
                              void* d_out, int out_size, void* d_ws, size_t ws_size,
                              hipStream_t stream) {
    const float* X    = (const float*)d_in[0];
    const float* Wqkv = (const float*)d_in[1];
    const float* Wout = (const float*)d_in[2];
    float* out = (float*)d_out;

    const size_t nQKV  = (size_t)B_ * H_ * S_ * DH_;   // 4,194,304
    const size_t nX    = (size_t)B_ * S_ * D_;         // 4,194,304
    const size_t nWq   = (size_t)3 * D_ * D_;          // 3,145,728
    const size_t nWo   = (size_t)D_ * D_;              // 1,048,576

    bf16* base = (bf16*)d_ws;
    bf16* Q    = base;               // 8 MiB
    bf16* K    = Q  + nQKV;          // 8 MiB
    bf16* Vt   = K  + nQKV;          // 8 MiB
    bf16* vals = Vt + nQKV;          // 8 MiB
    bf16* Xbf  = vals + nQKV;        // 8 MiB
    bf16* Wqb  = Xbf + nX;           // 6 MiB
    bf16* Wob  = Wqb + nWq;          // 2 MiB   (total ~50 MiB)

    // convert inputs to bf16 once
    cvt_kernel<<<(int)(nX  / 8 / 256), 256, 0, stream>>>(X,    Xbf, (int)(nX  / 8));
    cvt_kernel<<<(int)(nWq / 8 / 256), 256, 0, stream>>>(Wqkv, Wqb, (int)(nWq / 8));
    cvt_kernel<<<(int)(nWo / 8 / 256), 256, 0, stream>>>(Wout, Wob, (int)(nWo / 8));

    // QKV GEMM: 128 m-tiles(32) * 48 n-strips(64) = 6144 waves / 4 per block
    qkv_kernel<<<1536, 128, 0, stream>>>(Xbf, Wqb, Q, K, Vt);
    // attention: B*H*(S/32) = 2048 query tiles / 4 waves per block
    attn_kernel<<<512, 128, 0, stream>>>(Q, K, Vt, vals);
    // out GEMM: 128 m-tiles(32) * 16 n-strips(64) = 2048 waves / 4 per block
    out_kernel<<<512, 128, 0, stream>>>(vals, Wob, out);
}
